// LSTM_46377056862852
// MI455X (gfx1250) — matrix operand order repaired and checked by reading, not compiled, not run on hardware
//
#include <hip/hip_runtime.h>
#include <hip/hip_bf16.h>

namespace {

constexpr int TT  = 512;          // time steps
constexpr int BB  = 64;           // batch
constexpr int HH  = 1024;         // hidden
constexpr int LL  = 3;            // layers
constexpr int FHH = 4 * HH;       // 4096 (i,f,g,o)
constexpr float EPSV = 1e-5f;

constexpr int MTILES = BB / 16;   // 4   M tiles (batch)
constexpr int KTILES = HH / 32;   // 32  K tiles
constexpr int NTILES = FHH / 16;  // 256 N tiles
constexpr int FRAG   = 512;       // halfs per 16x32 (A) or 32x16 (B) fragment

constexpr size_t SEQ_HALFS   = (size_t)TT * MTILES * KTILES * FRAG;  // 33,554,432
constexpr size_t WPACK_HALFS = (size_t)NTILES * KTILES * FRAG;       // 4,194,304 / layer
constexpr size_t HPACK_HALFS = (size_t)MTILES * KTILES * FRAG;       // 65,536

typedef __attribute__((ext_vector_type(16))) __bf16 v16bf;
typedef __attribute__((ext_vector_type(8)))  float  v8f;

// ---- WMMA fragment element positions (ISA 7.12.2, 16-bit layouts) ----
// A (16x32, MxK): lane = ((kk>>3)&1)*16 + m ; half index = ((kk>>4)<<3)|(kk&7)
__device__ __forceinline__ int a_pos(int m, int kk) {
  int h    = ((kk >> 4) << 3) | (kk & 7);
  int lane = (((kk >> 3) & 1) << 4) | m;
  return lane * 16 + h;
}
// B (32x16, KxN): lane = ((kk>>3)&1)*16 + n ; half index = ((kk>>4)<<3)|(kk&7)
__device__ __forceinline__ int b_pos(int nn, int kk) {
  int lane = (((kk >> 3) & 1) << 4) | nn;
  return lane * 16 + (((kk >> 4) << 3) | (kk & 7));
}

// ---- pack x[t,b,k] (fp32) -> per-timestep A fragments (bf16) ----
__global__ void pack_x_kernel(const float* __restrict__ x, __bf16* __restrict__ seq) {
  size_t i = (size_t)blockIdx.x * blockDim.x + threadIdx.x;
  if (i >= (size_t)TT * BB * HH) return;
  int k = (int)(i % HH);
  int b = (int)((i / HH) % BB);
  int t = (int)(i / ((size_t)BB * HH));
  size_t blk = ((size_t)t * MTILES + (b >> 4)) * KTILES + (k >> 5);
  seq[blk * FRAG + a_pos(b & 15, k & 31)] = (__bf16)x[i];
}

// ---- pack Wx/Wh[l,n,k] (fp32) -> B fragments (bf16), all layers ----
__global__ void pack_w_kernel(const float* __restrict__ Wx, const float* __restrict__ Wh,
                              __bf16* __restrict__ Wxp, __bf16* __restrict__ Whp) {
  size_t i = (size_t)blockIdx.x * blockDim.x + threadIdx.x;
  if (i >= (size_t)LL * FHH * HH) return;
  int k = (int)(i % HH);
  int n = (int)((i / HH) % FHH);
  int l = (int)(i / ((size_t)FHH * HH));
  size_t blk = (size_t)(n >> 4) * KTILES + (k >> 5);
  size_t dst = (size_t)l * WPACK_HALFS + blk * FRAG + b_pos(n & 15, k & 31);
  Wxp[dst] = (__bf16)Wx[i];
  Whp[dst] = (__bf16)Wh[i];
}

// ---- zero h/c state + sync counters at layer start ----
__global__ void init_kernel(__bf16* hpack, float* cstate, unsigned* sync) {
  int i = blockIdx.x * blockDim.x + threadIdx.x;
  if (i < (int)HPACK_HALFS) hpack[i] = (__bf16)0.0f;
  if (i < BB * HH)          cstate[i] = 0.0f;
  if (i == 0) { sync[0] = 0u; sync[1] = 0u; }
}

// ---- software grid barrier (arrive counter + generation spin) ----
__device__ __forceinline__ void grid_sync(unsigned* sync, unsigned nwg) {
  __syncthreads();
  if (threadIdx.x == 0) {
    __threadfence();
    volatile unsigned* gen = sync + 1;
    unsigned g = *gen;
    if (atomicAdd(sync, 1u) == nwg - 1u) {
      sync[0] = 0u;
      __threadfence();
      atomicAdd((unsigned*)(sync + 1), 1u);
    } else {
      while (*gen == g) __builtin_amdgcn_s_sleep(1);
    }
    __threadfence();
  }
  __syncthreads();
}

// ---- sum + sumsq reduction over 256 threads: wave32 butterfly + 8-partial
// LDS combine. Exactly one __syncthreads; `slot` = 16 disjoint floats. ----
__device__ __forceinline__ void wg_reduce2(float& s, float& s2, volatile float* slot) {
  #pragma unroll
  for (int m = 16; m > 0; m >>= 1) {
    s  += __shfl_xor(s,  m, 32);
    s2 += __shfl_xor(s2, m, 32);
  }
  int wid = threadIdx.x >> 5;
  if ((threadIdx.x & 31) == 0) { slot[wid * 2] = s; slot[wid * 2 + 1] = s2; }
  __syncthreads();
  s = 0.f; s2 = 0.f;
  #pragma unroll
  for (int w = 0; w < 8; ++w) { s += slot[w * 2]; s2 += slot[w * 2 + 1]; }
}

// ---- persistent per-layer recurrent kernel ----
// grid = 128 WGs x 256 thr = 1024 waves; phase1: 1 ifgo 16x16 tile per wave
// with two independent WMMA accumulator chains (hides TRANS32 dep latency);
// phase2: WG b<64 does LSTM cell + 5 layernorms for batch row b.
__global__ void __launch_bounds__(256)
lstm_layer_kernel(const __bf16* __restrict__ seqin,
                  __bf16* __restrict__ seqout,
                  const __bf16* __restrict__ Wxp,
                  const __bf16* __restrict__ Whp,
                  const float* __restrict__ bh,     // [4096]
                  const float* __restrict__ ln_g,   // [4,H]
                  const float* __restrict__ ln_b,   // [4,H]
                  const float* __restrict__ lnc_g,  // [H]
                  const float* __restrict__ lnc_b,  // [H]
                  __bf16* __restrict__ hpack,       // [MTILES*KTILES*FRAG]
                  float* __restrict__ cstate,       // [B,H]
                  float* __restrict__ ifgo,         // [B,4096]
                  float* __restrict__ out_seq,      // [T,B,H] (last layer)
                  float* __restrict__ hfin,         // [B,H]
                  float* __restrict__ cfin,         // [B,H]
                  unsigned* sync, int is_last)
{
  __shared__ float red[5 * 16];             // 5 disjoint reduction slots
  const int tid  = threadIdx.x;
  const int lane = tid & 31;
  const int wave = tid >> 5;
  const int tile = blockIdx.x * 8 + wave;   // 0..1023
  const int mt   = tile >> 8;               // 0..3
  const int nt   = tile & 255;              // 0..255
  const unsigned NWG = gridDim.x;

  for (int t = 0; t < TT; ++t) {
    // ===== Phase 1: ifgo = x_t @ Wx^T + h @ Wh^T  (bf16 WMMA, f32 acc) =====
    v8f acc0 = {}, acc1 = {};
    {
      const v16bf* ax = (const v16bf*)(seqin + ((size_t)t * MTILES + mt) * KTILES * FRAG) + lane;
      const v16bf* bx = (const v16bf*)(Wxp + (size_t)nt * KTILES * FRAG) + lane;
      #pragma unroll 2
      for (int kt = 0; kt < KTILES; kt += 2) {
        v16bf a0 = ax[kt * 32];
        v16bf b0 = bx[kt * 32];
        v16bf a1 = ax[(kt + 1) * 32];
        v16bf b1 = bx[(kt + 1) * 32];
        acc0 = __builtin_amdgcn_wmma_f32_16x16x32_bf16(false, a0, false, b0, (short)0, acc0, false, false);
        acc1 = __builtin_amdgcn_wmma_f32_16x16x32_bf16(false, a1, false, b1, (short)0, acc1, false, false);
      }
      const v16bf* ah = (const v16bf*)(hpack + (size_t)mt * KTILES * FRAG) + lane;
      const v16bf* bw = (const v16bf*)(Whp + (size_t)nt * KTILES * FRAG) + lane;
      #pragma unroll 2
      for (int kt = 0; kt < KTILES; kt += 2) {
        v16bf a0 = ah[kt * 32];
        v16bf b0 = bw[kt * 32];
        v16bf a1 = ah[(kt + 1) * 32];
        v16bf b1 = bw[(kt + 1) * 32];
        acc0 = __builtin_amdgcn_wmma_f32_16x16x32_bf16(false, a0, false, b0, (short)0, acc0, false, false);
        acc1 = __builtin_amdgcn_wmma_f32_16x16x32_bf16(false, a1, false, b1, (short)0, acc1, false, false);
      }
    }
    {
      v8f acc = acc0 + acc1;
      // C/D layout: VGPR r, lane L -> row = r + (L/16)*8, col = L%16
      int m0 = mt * 16 + ((lane >> 4) << 3);
      int n  = nt * 16 + (lane & 15);
      #pragma unroll
      for (int r = 0; r < 8; ++r)
        ifgo[(size_t)(m0 + r) * FHH + n] = acc[r];
    }
    grid_sync(sync, NWG);

    // ===== Phase 2: LSTM cell, one WG per batch row =====
    if (blockIdx.x < (unsigned)BB) {
      const int b = blockIdx.x;
      float g4[4][4];   // [gate][q], col j = tid + q*256
      float mean[4], rstd[4];
      #pragma unroll
      for (int gidx = 0; gidx < 4; ++gidx) {
        float s = 0.f, s2 = 0.f;
        #pragma unroll
        for (int q = 0; q < 4; ++q) {
          int j = tid + q * 256;
          float v = ifgo[(size_t)b * FHH + gidx * HH + j] + bh[gidx * HH + j];
          g4[gidx][q] = v; s += v; s2 += v * v;
        }
        wg_reduce2(s, s2, red + gidx * 16);
        mean[gidx] = s * (1.0f / HH);
        rstd[gidx] = rsqrtf(s2 * (1.0f / HH) - mean[gidx] * mean[gidx] + EPSV);
      }
      #pragma unroll
      for (int gidx = 0; gidx < 4; ++gidx)
        #pragma unroll
        for (int q = 0; q < 4; ++q) {
          int j = tid + q * 256;
          g4[gidx][q] = (g4[gidx][q] - mean[gidx]) * rstd[gidx] * ln_g[gidx * HH + j] + ln_b[gidx * HH + j];
        }
      // c_next = sig(f)*c + sig(i)*tanh(g)
      float cn[4];
      float s = 0.f, s2 = 0.f;
      #pragma unroll
      for (int q = 0; q < 4; ++q) {
        int j = tid + q * 256;
        float ig = 1.f / (1.f + __expf(-g4[0][q]));
        float fg = 1.f / (1.f + __expf(-g4[1][q]));
        float c  = fg * cstate[b * HH + j] + ig * tanhf(g4[2][q]);
        cn[q] = c; s += c; s2 += c * c;
      }
      wg_reduce2(s, s2, red + 4 * 16);
      float cmean = s * (1.0f / HH);
      float crstd = rsqrtf(s2 * (1.0f / HH) - cmean * cmean + EPSV);
      #pragma unroll
      for (int q = 0; q < 4; ++q) {
        int j = tid + q * 256;
        float c  = cn[q];
        float og = 1.f / (1.f + __expf(-g4[3][q]));
        float hv = og * tanhf((c - cmean) * crstd * lnc_g[j] + lnc_b[j]);
        cstate[b * HH + j] = c;
        int pos    = a_pos(b & 15, j & 31);
        size_t blk = (size_t)(b >> 4) * KTILES + (j >> 5);
        hpack[blk * FRAG + pos] = (__bf16)hv;          // next step's A fragments
        if (!is_last) {
          size_t sblk = ((size_t)t * MTILES + (b >> 4)) * KTILES + (j >> 5);
          seqout[sblk * FRAG + pos] = (__bf16)hv;      // next layer's input frags
        } else {
          out_seq[((size_t)t * BB + b) * HH + j] = hv; // final output sequence
        }
        if (t == TT - 1) {
          hfin[b * HH + j] = hv;
          cfin[b * HH + j] = c;
        }
      }
    }
    grid_sync(sync, NWG);
  }
}

} // anonymous namespace

extern "C" void kernel_launch(void* const* d_in, const int* in_sizes, int n_in,
                              void* d_out, int out_size, void* d_ws, size_t ws_size,
                              hipStream_t stream) {
  (void)in_sizes; (void)n_in; (void)out_size; (void)ws_size;
  const float* x     = (const float*)d_in[0];
  const float* Wx    = (const float*)d_in[1];
  const float* Wh    = (const float*)d_in[2];
  const float* bh    = (const float*)d_in[3];
  const float* ln_g  = (const float*)d_in[4];
  const float* ln_b  = (const float*)d_in[5];
  const float* lnc_g = (const float*)d_in[6];
  const float* lnc_b = (const float*)d_in[7];
  float* out = (float*)d_out;   // [T*B*H] out ++ [L*B*H] h ++ [L*B*H] c

  // workspace carve-up (~186 MB total)
  char* ws = (char*)d_ws;
  auto carve = [&](size_t bytes) { char* p = ws; ws += (bytes + 255) & ~(size_t)255; return p; };
  __bf16*  seqA   = (__bf16*)carve(SEQ_HALFS * 2);
  __bf16*  seqB   = (__bf16*)carve(SEQ_HALFS * 2);
  __bf16*  Wxp    = (__bf16*)carve((size_t)LL * WPACK_HALFS * 2);
  __bf16*  Whp    = (__bf16*)carve((size_t)LL * WPACK_HALFS * 2);
  __bf16*  hpack  = (__bf16*)carve(HPACK_HALFS * 2);
  float*   cstate = (float*)carve((size_t)BB * HH * 4);
  float*   ifgo   = (float*)carve((size_t)BB * FHH * 4);
  unsigned* sync  = (unsigned*)carve(256);

  pack_x_kernel<<<(TT * BB * HH) / 256, 256, 0, stream>>>(x, seqA);
  pack_w_kernel<<<(LL * FHH * HH) / 256, 256, 0, stream>>>(Wx, Wh, Wxp, Whp);

  __bf16* sin = seqA;
  __bf16* sout = seqB;
  for (int l = 0; l < LL; ++l) {
    init_kernel<<<(int)(HPACK_HALFS / 256), 256, 0, stream>>>(hpack, cstate, sync);
    int is_last = (l == LL - 1);
    lstm_layer_kernel<<<128, 256, 0, stream>>>(
        sin, sout,
        Wxp + (size_t)l * WPACK_HALFS, Whp + (size_t)l * WPACK_HALFS,
        bh + (size_t)l * FHH,
        ln_g + (size_t)l * 4 * HH, ln_b + (size_t)l * 4 * HH,
        lnc_g + (size_t)l * HH,    lnc_b + (size_t)l * HH,
        hpack, cstate, ifgo,
        out,
        out + (size_t)TT * BB * HH + (size_t)l * BB * HH,
        out + (size_t)TT * BB * HH + (size_t)LL * BB * HH + (size_t)l * BB * HH,
        sync, is_last);
    __bf16* tmp = sin; sin = sout; sout = tmp;
  }
}
